// NodeModel_29137058136337
// MI455X (gfx1250) — matrix operand order repaired and checked by reading, not compiled, not run on hardware
//
#include <hip/hip_runtime.h>

typedef __attribute__((ext_vector_type(16))) _Float16 v16h;
typedef __attribute__((ext_vector_type(8)))  _Float16 v8h;
typedef __attribute__((ext_vector_type(8)))  float    v8f;

#define NNODES 50000
#define NEDGES 800000
#define IN_C   128
#define MSG_C  64
#define OUT_C  128
// total weight elements: 192*128 + 128*128 + 128*128 = 57344
#define WTOT   57344
#define HPAD   200   // padded row length (halfs); 200*2=400B, 16B aligned per row

// ---------------------------------------------------------------------------
// Kernel 1: scatter-add messages into per-node aggregation buffer
// one thread per (edge, channel) -> coalesced message reads, f32 atomics
// ---------------------------------------------------------------------------
__global__ void scatter_kernel(const float* __restrict__ msg,
                               const int*   __restrict__ edge_index,
                               float*       __restrict__ aggr)
{
    unsigned t = blockIdx.x * 256u + threadIdx.x;           // E*64 = 51.2M
    if (t >= (unsigned)NEDGES * MSG_C) return;
    unsigned e = t >> 6;
    unsigned c = t & 63u;
    int dest = edge_index[NEDGES + e];                      // row 1 = dest
    atomicAdd(&aggr[(size_t)dest * MSG_C + c], msg[t]);
}

// ---------------------------------------------------------------------------
// Kernel 2: convert weights f32 -> f16 in WMMA B-fragment order.
// Fragment f (0..111): W1 f=[0,48), W2 [48,80), W3 [80,112). Within a frag:
// element index r = lane*16 + i ; lane<16 holds K=kc*32+i (i=0..7), K=kc*32+16+(i-8) (i=8..15),
// lane>=16 same with K shifted by 8 ; N = nt*16 + lane%16  (B is KxN row-major)
// ---------------------------------------------------------------------------
__global__ void wswizzle_kernel(const float* __restrict__ W1,
                                const float* __restrict__ W2,
                                const float* __restrict__ W3,
                                _Float16*    __restrict__ wfrag)
{
    int t = blockIdx.x * 256 + threadIdx.x;
    if (t >= WTOT) return;
    int f    = t >> 9;          // fragment id (512 elements each)
    int r    = t & 511;
    int lane = r >> 4;
    int i    = r & 15;
    const float* W;
    int kc;
    if (f < 48)      { W = W1; kc = f >> 3; }
    else if (f < 80) { W = W2; kc = (f - 48) >> 3; }
    else             { W = W3; kc = (f - 80) >> 3; }
    int nt = f & 7;
    int k  = kc * 32 + (lane >> 4) * 8 + (i & 7) + (i >> 3) * 16;
    int n  = nt * 16 + (lane & 15);
    wfrag[t] = (_Float16)W[k * OUT_C + n];
}

// ---------------------------------------------------------------------------
// A-fragment load from LDS row-major f16 tile (ISA 7.12.2 16-bit A layout):
// lane<16: M=lane, elems 0..7 -> K=kbase..kbase+7, elems 8..15 -> K=kbase+16..+23
// lane>=16: same M, kbase shifted by 8.
// ---------------------------------------------------------------------------
__device__ inline v16h load_afrag(const _Float16* hrow, int kc, int lhalf)
{
    int kbase = kc * 32 + lhalf * 8;
    v8h lo = *(const v8h*)(hrow + kbase);
    v8h hi = *(const v8h*)(hrow + kbase + 16);
    v16h a;
#pragma unroll
    for (int i = 0; i < 8; ++i) { a[i] = lo[i]; a[i + 8] = hi[i]; }
    return a;
}

__device__ inline v16h load_bfrag(const _Float16* wlds, int f, int lane)
{
    return *(const v16h*)(wlds + f * 512 + lane * 16);
}

// Layers 1 & 2: GEMM + bias + leaky_relu(0.2), result back into LDS tile cols 0..127
template <int NK>
__device__ inline void layer_fwd(_Float16 (*h)[HPAD], const _Float16* wlds, int fbase,
                                 const float* __restrict__ bias, int lhalf, int lmod, int lane)
{
    v16h afr[NK];
#pragma unroll
    for (int kc = 0; kc < NK; ++kc) afr[kc] = load_afrag(&h[lmod][0], kc, lhalf);

#pragma unroll
    for (int nt = 0; nt < 8; ++nt) {
        v8f acc = {};
#pragma unroll
        for (int kc = 0; kc < NK; ++kc) {
            v16h b = load_bfrag(wlds, fbase + kc * 8 + nt, lane);
            acc = __builtin_amdgcn_wmma_f32_16x16x32_f16(
                false, afr[kc], false, b, (short)0, acc, false, false);
        }
        float bv = bias[nt * 16 + lmod];
#pragma unroll
        for (int r = 0; r < 8; ++r) {
            float v = acc[r] + bv;
            v = v > 0.f ? v : 0.2f * v;               // leaky relu, slope 0.2
            h[lhalf * 8 + r][nt * 16 + lmod] = (_Float16)v;  // C layout: M=r+8*lhalf, N=lmod
        }
    }
}

// ---------------------------------------------------------------------------
// Kernel 3: fused MLP + LayerNorm. 128 threads = 4 waves, 16 nodes per wave.
// Weight tile is DMA'd into LDS with CDNA5 async global->LDS copies
// (no VGPR roundtrip), overlapped with f32->f16 activation staging.
// ---------------------------------------------------------------------------
__launch_bounds__(128, 1)
__global__ void mlp_kernel(const float* __restrict__ x,
                           const float* __restrict__ aggr,
                           const _Float16* __restrict__ wfrag,
                           const float* __restrict__ b1,
                           const float* __restrict__ b2,
                           const float* __restrict__ gamma,
                           const float* __restrict__ beta,
                           float* __restrict__ out)
{
    __shared__ __align__(32) _Float16 wlds[WTOT];        // 112 KB, f16 B-fragments
    __shared__ __align__(16) _Float16 hlds[4][16][HPAD]; // 4 x 6.25 KB activation tiles

    const int tid   = threadIdx.x;
    const int wave  = tid >> 5;
    const int lane  = tid & 31;
    const int lhalf = lane >> 4;
    const int lmod  = lane & 15;
    const int rowbase = blockIdx.x * 64 + wave * 16;

    // --- issue async weight DMA: global -> LDS, 16B per lane per pass ---
    // 114688 B total / (128 threads * 16 B) = 56 passes. ASYNCcnt-tracked;
    // hardware back-pressures issue if the 6-bit counter would overflow.
    {
        unsigned lds_base = (unsigned)(uintptr_t)&wlds[0]; // low 32 bits = LDS offset
        const void* gbase = (const void*)wfrag;
#pragma unroll 4
        for (int it = 0; it < 56; ++it) {
            unsigned off  = (unsigned)(tid * 16 + it * 2048);
            unsigned ldsa = lds_base + off;
            asm volatile("global_load_async_to_lds_b128 %0, %1, %2"
                         :: "v"(ldsa), "v"(off), "s"(gbase)
                         : "memory");
        }
    }

    // --- stage h0 = [x | msg_aggr] as f16 into this wave's tile (overlaps DMA) ---
    _Float16 (*h)[HPAD] = hlds[wave];
#pragma unroll
    for (int r = 0; r < 16; ++r) {
        int grow = rowbase + r;
        for (int c = lane; c < 192; c += 32) {
            float v = 0.f;
            if (grow < NNODES)
                v = (c < IN_C) ? x[(size_t)grow * IN_C + c]
                               : aggr[(size_t)grow * MSG_C + (c - IN_C)];
            h[r][c] = (_Float16)v;
        }
    }

    // drain this wave's async copies, then block-sync so all waves see wlds
    asm volatile("s_wait_asynccnt 0x0" ::: "memory");
    __syncthreads();

    // --- layer 1: [16x192] @ W1 -> [16x128], leaky relu ---
    layer_fwd<6>(h, wlds, /*fbase=*/0,  b1, lhalf, lmod, lane);
    // --- layer 2: [16x128] @ W2 -> [16x128], leaky relu ---
    layer_fwd<4>(h, wlds, /*fbase=*/48, b2, lhalf, lmod, lane);

    // --- layer 3: keep all 8 C tiles in registers, then LayerNorm in-regs ---
    v16h afr[4];
#pragma unroll
    for (int kc = 0; kc < 4; ++kc) afr[kc] = load_afrag(&h[lmod][0], kc, lhalf);

    v8f accs[8];
#pragma unroll
    for (int nt = 0; nt < 8; ++nt) {
        v8f acc = {};
#pragma unroll
        for (int kc = 0; kc < 4; ++kc) {
            v16h b = load_bfrag(wlds, 80 + kc * 8 + nt, lane);
            acc = __builtin_amdgcn_wmma_f32_16x16x32_f16(
                false, afr[kc], false, b, (short)0, acc, false, false);
        }
        accs[nt] = acc;
    }

    float gl[8], bl[8];
#pragma unroll
    for (int nt = 0; nt < 8; ++nt) {
        gl[nt] = gamma[nt * 16 + lmod];
        bl[nt] = beta[nt * 16 + lmod];
    }

    // Row M = lhalf*8 + r lives across the 16 lanes of this half-wave
    // (col n = nt*16 + lmod). Reduce with xor-shuffles that stay inside the half.
#pragma unroll
    for (int r = 0; r < 8; ++r) {
        float s = 0.f, q = 0.f;
#pragma unroll
        for (int nt = 0; nt < 8; ++nt) {
            float v = accs[nt][r];
            s += v; q += v * v;
        }
#pragma unroll
        for (int m = 1; m < 16; m <<= 1) {
            s += __shfl_xor(s, m, 32);
            q += __shfl_xor(q, m, 32);
        }
        float mu   = s * (1.f / 128.f);
        float var  = q * (1.f / 128.f) - mu * mu;
        float rstd = rsqrtf(var + 1e-5f);
        int grow = rowbase + lhalf * 8 + r;
        if (grow < NNODES) {
#pragma unroll
            for (int nt = 0; nt < 8; ++nt)
                out[(size_t)grow * OUT_C + nt * 16 + lmod] =
                    (accs[nt][r] - mu) * rstd * gl[nt] + bl[nt];
        }
    }
}

// ---------------------------------------------------------------------------
extern "C" void kernel_launch(void* const* d_in, const int* in_sizes, int n_in,
                              void* d_out, int out_size, void* d_ws, size_t ws_size,
                              hipStream_t stream)
{
    const float* x     = (const float*)d_in[0];
    const int*   ei    = (const int*)  d_in[1];   // edge_index [2,E]
    const float* msg   = (const float*)d_in[2];
    const float* W1    = (const float*)d_in[3];
    const float* b1    = (const float*)d_in[4];
    const float* W2    = (const float*)d_in[5];
    const float* b2    = (const float*)d_in[6];
    const float* W3    = (const float*)d_in[7];
    const float* gamma = (const float*)d_in[8];
    const float* beta  = (const float*)d_in[9];
    float* out = (float*)d_out;

    // workspace: [0, N*64*4) = msg_aggr f32 ; then f16 weight fragments
    float*    aggr  = (float*)d_ws;
    _Float16* wfrag = (_Float16*)((char*)d_ws + (size_t)NNODES * MSG_C * sizeof(float));

    hipMemsetAsync(aggr, 0, (size_t)NNODES * MSG_C * sizeof(float), stream);

    wswizzle_kernel<<<(WTOT + 255) / 256, 256, 0, stream>>>(W1, W2, W3, wfrag);

    unsigned scat_threads = (unsigned)NEDGES * MSG_C;
    scatter_kernel<<<(scat_threads + 255) / 256, 256, 0, stream>>>(msg, ei, aggr);

    mlp_kernel<<<(NNODES + 63) / 64, 128, 0, stream>>>(x, aggr, wfrag,
                                                       b1, b2, gamma, beta, out);
}
